// CaC_Module_43361989820672
// MI455X (gfx1250) — matrix operand order repaired and checked by reading, not compiled, not run on hardware
//
#include <hip/hip_runtime.h>
#include <hip/hip_bf16.h>
#include <math.h>

// ---------------------------------------------------------------------------
// CDNA5 (gfx1250, wave32) implementation of the dynamic-depthwise-conv module.
//   key  = Wk @ x + bk          -> bf16 WMMA GEMM (dominant 17.2 GFLOP)
//                                  B-tile staged via Tensor Data Mover (TDM)
//   qry  = softmax(Wq @ x + bq) -> small VALU GEMM + row softmax
//   kern = key @ qry^T          -> wave-parallel dot-products (L2-resident key)
//   out  = x * sum_d sigmoid(depthwise3x3(x, kern, d)), d in {1,3,5}
// ---------------------------------------------------------------------------

typedef __bf16 bf16_t;
typedef bf16_t v16bf __attribute__((ext_vector_type(16)));
typedef float  v8f   __attribute__((ext_vector_type(8)));
typedef unsigned int v4u __attribute__((ext_vector_type(4)));
typedef int v8i __attribute__((ext_vector_type(8)));
typedef int v4i __attribute__((ext_vector_type(4)));

#define BB 8
#define CC 512
#define HH 64
#define WW 64
#define HWN 4096
#define K9 9

// ------------------------- Kernel 1: key GEMM (WMMA) -----------------------
// Per batch: A = Wk [512 x 512], B = x[b] [512 x 4096] -> key[b] [512 x 4096].
// Block tile M=128, N=64, K-step 32. 256 threads = 8 waves (2 M-waves x 4 N-waves).
// Each wave computes a 64(M) x 16(N) strip = 4 subtiles of 16x16 via
// v_wmma_f32_16x16x32_bf16 (f32 accumulate). B tile is DMA'd by the TDM
// (tensor_load_to_lds, tracked by TENSORcnt), then converted f32->bf16 in LDS.
#define TM 128
#define TN 64
#define TK 32
#define LDA 40   // LDS K-stride (bf16 elems) for A tile [m][k], padded vs bank conflicts
#define LDB 40   // LDS K-stride for bf16 B tile stored n-major [n][k]

__global__ __launch_bounds__(256)
void gemm_key_kernel(const float* __restrict__ x, const float* __restrict__ Wk,
                     const float* __restrict__ bk, float* __restrict__ key)
{
    __shared__ bf16_t As[TM * LDA];   // [m][k] bf16
    __shared__ float  Bsf[TK * TN];   // raw f32 tile [k][n], filled by TDM
    __shared__ bf16_t Bs[TN * LDB];   // [n][k] bf16 (fragment-ready layout)

    const int n0 = blockIdx.x * TN;
    const int m0 = blockIdx.y * TM;
    const int b  = blockIdx.z;
    const float* __restrict__ xb   = x   + (size_t)b * CC * HWN;
    float*       __restrict__ keyb = key + (size_t)b * CC * HWN;

    const int tid  = threadIdx.x;
    const int lane = tid & 31;
    const int wid  = tid >> 5;
    const int wm   = wid >> 2;     // 0..1  (M wave)
    const int wn   = wid & 3;      // 0..3  (N wave)
    const int fn   = lane & 15;    // element index 0..15 within subtile row
    const int half = lane >> 4;    // lane half (0/1)

    v8f acc[4] = {};               // 4 M-subtiles of 16x16 f32

    // Invariant parts of the Tensor DMA Descriptor (D#), per ISA 08_async_tensor §8.
    const unsigned int lds_off = (unsigned int)(uintptr_t)(void*)Bsf;

    for (int k0 = 0; k0 < CC; k0 += TK) {
        // ---- wave 0: issue TDM load of B tile (32 k-rows x 64 n-cols, f32) ----
        if (wid == 0) {
            const unsigned long long ga =
                (unsigned long long)(uintptr_t)(xb + (size_t)k0 * HWN + n0);
            v4u g0;
            g0[0] = 1u;                                   // count=1, user mode
            g0[1] = lds_off;                              // lds_addr [63:32]
            g0[2] = (unsigned int)ga;                     // global_addr [95:64]
            g0[3] = (unsigned int)((ga >> 32) & 0x1FFFFFFu) | (2u << 30); // [120:96], type=2
            v8i g1;
            g1[0] = (2 << 16);                            // wg_mask=0, data_size=4B
            g1[1] = (int)((HWN & 0xFFFF) << 16);          // tensor_dim0[15:0]  (=4096)
            g1[2] = (int)((HWN >> 16) | ((CC & 0xFFFF) << 16)); // dim0 hi | tensor_dim1 lo (=512)
            g1[3] = (int)((CC >> 16) | (TN << 16));       // dim1 hi | tile_dim0 (=64)
            g1[4] = TK;                                   // tile_dim1 (=32), tile_dim2=0
            g1[5] = HWN;                                  // tensor_dim0_stride lo (=4096)
            g1[6] = 0;                                    // stride0 hi | stride1 lo
            g1[7] = 0;                                    // stride1 hi
            v4i gz = (v4i)0;
#if defined(__clang_major__) && (__clang_major__ >= 23)
            v8i gz8 = (v8i)0;
            __builtin_amdgcn_tensor_load_to_lds(g0, g1, gz, gz, gz8, 0);
#else
            __builtin_amdgcn_tensor_load_to_lds(g0, g1, gz, gz, 0);
#endif
            __builtin_amdgcn_s_wait_tensorcnt(0);
        }
        // ---- all waves: stage A tile (Wk): 128 rows x 32 k, f32 -> bf16 ----
        {
            const int row = tid >> 1;
            const int cb  = (tid & 1) * 16;
            const float* src = Wk + (size_t)(m0 + row) * CC + k0 + cb;
            bf16_t* dst = &As[row * LDA + cb];
            #pragma unroll
            for (int i = 0; i < 16; ++i) dst[i] = (bf16_t)src[i];
        }
        if (k0 + TK < CC)   // prefetch next A source (global_prefetch_b8)
            __builtin_prefetch(Wk + (size_t)(m0 + (tid >> 1)) * CC + k0 + TK, 0, 0);
        __syncthreads();    // TDM tile + A tile visible

        // ---- convert B tile f32 [k][n] -> bf16 [n][k] (fragment-ready) ----
        {
            const int row = tid >> 3;           // k 0..31
            const int cb  = (tid & 7) * 8;      // n base
            #pragma unroll
            for (int i = 0; i < 8; ++i)
                Bs[(cb + i) * LDB + row] = (bf16_t)Bsf[row * TN + cb + i];
        }
        __syncthreads();

        // ---- B fragment (32x16 bf16): lane -> n = wn*16+fn, k pairs per VGPR:
        //      k = 16*half + 2v, 2v+1  (pairs contiguous in LDS due to [n][k] layout)
        v16bf bfrag;
        {
            const bf16_t* bp = &Bs[(wn * 16 + fn) * LDB + half * 16];
            #pragma unroll
            for (int v = 0; v < 8; ++v) {
                bfrag[2 * v]     = bp[2 * v];
                bfrag[2 * v + 1] = bp[2 * v + 1];
            }
        }
        // ---- 4 M-subtiles: A fragment (16x32 bf16) in contiguous k order (same as B):
        //      m = lane&15 ; VGPR v holds k = 16*half + 2v , +1
        #pragma unroll
        for (int s = 0; s < 4; ++s) {
            const bf16_t* ap = &As[(wm * 64 + s * 16 + fn) * LDA];
            v16bf afrag;
            #pragma unroll
            for (int v = 0; v < 8; ++v) {
                const int kb = 16 * half + 2 * v;
                afrag[2 * v]     = ap[kb];
                afrag[2 * v + 1] = ap[kb + 1];
            }
            acc[s] = __builtin_amdgcn_wmma_f32_16x16x32_bf16(
                false, afrag, false, bfrag, (short)0, acc[s], false, false);
        }
        __syncthreads();
    }

    // ---- store D + bias. C/D layout: lane(fn,half), VGPR v -> m_local = v + 8*half
    #pragma unroll
    for (int s = 0; s < 4; ++s) {
        #pragma unroll
        for (int v = 0; v < 8; ++v) {
            const int m = m0 + wm * 64 + s * 16 + v + 8 * half;
            const int n = n0 + wn * 16 + fn;
            keyb[(size_t)m * HWN + n] = acc[s][v] + bk[m];
        }
    }
}

// ------------------- Kernel 2: query = Wq @ x + bq (9 rows) ----------------
__global__ __launch_bounds__(256)
void query_kernel(const float* __restrict__ x, const float* __restrict__ Wq,
                  const float* __restrict__ bq, float* __restrict__ qry)
{
    const int t = blockIdx.x * 256 + threadIdx.x;   // over b*hw
    const int b = t >> 12;
    const int n = t & (HWN - 1);
    const float* xb = x + (size_t)b * CC * HWN + n;
    float acc[K9];
    #pragma unroll
    for (int o = 0; o < K9; ++o) acc[o] = bq[o];
    for (int c = 0; c < CC; ++c) {
        const float xv = xb[(size_t)c * HWN];       // coalesced across lanes
        #pragma unroll
        for (int o = 0; o < K9; ++o) acc[o] = fmaf(Wq[o * CC + c], xv, acc[o]);
    }
    #pragma unroll
    for (int o = 0; o < K9; ++o) qry[((size_t)b * K9 + o) * HWN + n] = acc[o];
}

// ---------------- Kernel 3: row softmax over hw=4096, in place -------------
__global__ __launch_bounds__(256)
void softmax_kernel(float* __restrict__ qry)
{
    __shared__ float red[8];
    float* row = qry + (size_t)blockIdx.x * HWN;
    const int tid = threadIdx.x, lane = tid & 31, wid = tid >> 5;

    float m = -3.0e38f;
    for (int n = tid; n < HWN; n += 256) m = fmaxf(m, row[n]);
    #pragma unroll
    for (int off = 16; off; off >>= 1) m = fmaxf(m, __shfl_xor(m, off, 32));
    if (lane == 0) red[wid] = m;
    __syncthreads();
    float mm = red[0];
    #pragma unroll
    for (int i = 1; i < 8; ++i) mm = fmaxf(mm, red[i]);
    __syncthreads();

    float s = 0.f;
    for (int n = tid; n < HWN; n += 256) {
        const float e = __expf(row[n] - mm);
        row[n] = e;
        s += e;
    }
    #pragma unroll
    for (int off = 16; off; off >>= 1) s += __shfl_xor(s, off, 32);
    if (lane == 0) red[wid] = s;
    __syncthreads();
    float tot = 0.f;
    #pragma unroll
    for (int i = 0; i < 8; ++i) tot += red[i];
    const float inv = 1.0f / tot;
    for (int n = tid; n < HWN; n += 256) row[n] *= inv;
}

// -------- Kernel 4: kern[b][c][k] = sum_n key[b][c][n] * qry[b][k][n] ------
// One wave per channel, 8 channels per WG; key stream is L2-resident.
__global__ __launch_bounds__(256)
void kern_kernel(const float* __restrict__ key, const float* __restrict__ qry,
                 float* __restrict__ kern)
{
    const int wid = threadIdx.x >> 5, lane = threadIdx.x & 31;
    const int gc = blockIdx.x * 8 + wid;       // b*C + c
    const int b = gc >> 9;
    const float* kr = key + (size_t)gc * HWN;
    const float* qb = qry + (size_t)b * K9 * HWN;
    float acc[K9] = {};
    for (int n = lane; n < HWN; n += 32) {
        const float kv = kr[n];
        #pragma unroll
        for (int k = 0; k < K9; ++k) acc[k] = fmaf(kv, qb[(size_t)k * HWN + n], acc[k]);
    }
    #pragma unroll
    for (int k = 0; k < K9; ++k) {
        float v = acc[k];
        #pragma unroll
        for (int off = 16; off; off >>= 1) v += __shfl_xor(v, off, 32);
        if (lane == 0) kern[(size_t)gc * K9 + k] = v;
    }
}

// ---- Kernel 5: out = x * (sig(conv_d1) + sig(conv_d3) + sig(conv_d5)) -----
// 32x32 spatial tile + halo 5 staged in LDS (zero-padded "same" boundary).
#define TILE 32
#define HALO 5
#define LT (TILE + 2 * HALO)   // 42

__global__ __launch_bounds__(256)
void dynconv_kernel(const float* __restrict__ x, const float* __restrict__ kern,
                    float* __restrict__ out)
{
    __shared__ float xs[LT * LT];
    __shared__ float kw[K9];
    const int bc  = blockIdx.z;           // b*C + c
    const int ty0 = blockIdx.y * TILE;
    const int tx0 = blockIdx.x * TILE;
    const float* xp = x + (size_t)bc * HWN;
    const int tid = threadIdx.x;

    if (tid < K9) kw[tid] = kern[(size_t)bc * K9 + tid];
    for (int i = tid; i < LT * LT; i += 256) {
        const int ly = i / LT, lx = i - ly * LT;
        const int gy = ty0 + ly - HALO, gx = tx0 + lx - HALO;
        float v = 0.f;
        if (gy >= 0 && gy < HH && gx >= 0 && gx < WW) v = xp[gy * WW + gx];
        xs[i] = v;
    }
    __syncthreads();

    const int lx = tid & 15, ly = tid >> 4;   // 16x16 threads, 2x2 px each
    #pragma unroll
    for (int py = 0; py < 2; ++py) {
        #pragma unroll
        for (int px = 0; px < 2; ++px) {
            const int oy = ly * 2 + py, ox = lx * 2 + px;
            float wsum = 0.f;
            #pragma unroll
            for (int di = 0; di < 3; ++di) {
                const int d = (di == 0) ? 1 : (di == 1) ? 3 : 5;
                float s = 0.f;
                #pragma unroll
                for (int i = 0; i < 3; ++i)
                    #pragma unroll
                    for (int j = 0; j < 3; ++j)
                        s = fmaf(kw[i * 3 + j],
                                 xs[(oy + HALO + (i - 1) * d) * LT +
                                    (ox + HALO + (j - 1) * d)], s);
                wsum += 1.0f / (1.0f + __expf(-s));
            }
            const int gy = ty0 + oy, gx = tx0 + ox;
            out[(size_t)bc * HWN + gy * WW + gx] = xp[gy * WW + gx] * wsum;
        }
    }
}

// ---------------------------------------------------------------------------
extern "C" void kernel_launch(void* const* d_in, const int* in_sizes, int n_in,
                              void* d_out, int out_size, void* d_ws, size_t ws_size,
                              hipStream_t stream) {
    (void)in_sizes; (void)n_in; (void)out_size; (void)ws_size;
    const float* x  = (const float*)d_in[0];
    const float* Wk = (const float*)d_in[1];
    const float* bk = (const float*)d_in[2];
    const float* Wq = (const float*)d_in[3];
    const float* bq = (const float*)d_in[4];
    float* out = (float*)d_out;

    // workspace layout (floats): key[8*512*4096] | qry[8*9*4096] | kern[8*512*9]
    float* key  = (float*)d_ws;
    float* qry  = key + (size_t)BB * CC * HWN;
    float* kern = qry + (size_t)BB * K9 * HWN;

    gemm_key_kernel<<<dim3(HWN / TN, CC / TM, BB), 256, 0, stream>>>(x, Wk, bk, key);
    query_kernel  <<<dim3((BB * HWN) / 256), 256, 0, stream>>>(x, Wq, bq, qry);
    softmax_kernel<<<dim3(BB * K9), 256, 0, stream>>>(qry);
    kern_kernel   <<<dim3((BB * CC) / 8), 256, 0, stream>>>(key, qry, kern);
    dynconv_kernel<<<dim3(WW / TILE, HH / TILE, BB * CC), 256, 0, stream>>>(x, kern, out);
}
